// TopK_88656714924078
// MI455X (gfx1250) — matrix operand order repaired and checked by reading, not compiled, not run on hardware
//
#include <hip/hip_runtime.h>
#include <hip/hip_bf16.h>

// ---------------------------------------------------------------------------
// CDNA5 (gfx1250) implementation. Core GEMMs use v_wmma_f32_16x16x32_bf16
// (f32 -> bf16 conversion fused into LDS staging, f32 accumulate).
// Software-pipelined: global float4 loads for K-chunk k+1 are in flight while
// the WMMAs for chunk k execute; ping-pong LDS buffers, one barrier per chunk.
// ---------------------------------------------------------------------------

typedef __attribute__((ext_vector_type(16))) __bf16 v16bf;
typedef __attribute__((ext_vector_type(8)))  float  v8f;

union FragBF { v16bf v; unsigned u[8]; };

#if defined(__has_builtin)
#  if __has_builtin(__builtin_amdgcn_cvt_pk_bf16_f32)
#    define HAVE_PK_BF16 1
#  endif
#endif

__device__ __forceinline__ unsigned short f2bf(float f) {
  unsigned x = __float_as_uint(f);
  x += 0x7FFFu + ((x >> 16) & 1u);          // round-to-nearest-even
  return (unsigned short)(x >> 16);
}
__device__ __forceinline__ unsigned pack_bf2(float lo, float hi) {
#if defined(HAVE_PK_BF16)
  auto r = __builtin_amdgcn_cvt_pk_bf16_f32(lo, hi);   // v_cvt_pk_bf16_f32
  unsigned u; __builtin_memcpy(&u, &r, 4); return u;
#else
  return ((unsigned)f2bf(hi) << 16) | (unsigned)f2bf(lo);
#endif
}

#define MODE_PLAIN   0
#define MODE_SIGMOID 1   // out = sigmoid(acc + bias[col])
#define MODE_RELU    2   // out = relu(acc + bias[col])

// Block tile 128x64, 8 waves (4x2), each wave: 2x2 WMMA 16x16 tiles, K step 32.
__global__ void __launch_bounds__(256)
gemm_wmma_bf16(const float* __restrict__ A, const float* __restrict__ B,
               float* __restrict__ C, int Nc, int K,
               long sA, long sB, long sC,
               const float* __restrict__ bias, int mode) {
  __shared__ unsigned short lA[2][128][40];   // ping-pong 128x32 bf16 A tiles
  __shared__ unsigned short lBt[2][64][40];   // ping-pong B tiles, transposed [n][k]

  const int bz = blockIdx.z;
  const float* Ab = A + (long)bz * sA;
  const float* Bb = B + (long)bz * sB;
  float*       Cb = C + (long)bz * sC;

  const int m0   = blockIdx.y * 128;
  const int n0   = blockIdx.x * 64;
  const int tid  = threadIdx.x;
  const int lane = tid & 31;
  const int wave = tid >> 5;
  const int wm   = (wave >> 1) * 32;   // 0,32,64,96
  const int wn   = (wave & 1) * 32;    // 0,32
  const int half = lane >> 4;
  const int l16  = lane & 15;

  // Per-thread staging slices (fixed, divergence-free):
  //   A tile 128x32: 4 x float4 per thread;  B tile 32x64: 2 x float4 per thread.
  const int aR[4] = { (tid * 4 + 0)    >> 5, (tid * 4 + 1024) >> 5,
                      (tid * 4 + 2048) >> 5, (tid * 4 + 3072) >> 5 };
  const int aC    = (tid * 4) & 31;
  const int bR[2] = { (tid * 4) >> 6, (tid * 4 + 1024) >> 6 };
  const int bC    = (tid * 4) & 63;

  v8f acc[2][2];
#pragma unroll
  for (int i = 0; i < 2; ++i)
#pragma unroll
    for (int j = 0; j < 2; ++j)
#pragma unroll
      for (int r = 0; r < 8; ++r) acc[i][j][r] = 0.0f;

  float4 ra[4], rb[2];

  // prologue: load + stage K-chunk 0
#pragma unroll
  for (int it = 0; it < 4; ++it)
    ra[it] = *(const float4*)&Ab[(long)(m0 + aR[it]) * K + aC];
#pragma unroll
  for (int it = 0; it < 2; ++it)
    rb[it] = *(const float4*)&Bb[(long)bR[it] * Nc + (n0 + bC)];

#pragma unroll
  for (int it = 0; it < 4; ++it) {
    uint2 u; u.x = pack_bf2(ra[it].x, ra[it].y); u.y = pack_bf2(ra[it].z, ra[it].w);
    *(uint2*)&lA[0][aR[it]][aC] = u;
  }
#pragma unroll
  for (int it = 0; it < 2; ++it) {
    lBt[0][bC + 0][bR[it]] = f2bf(rb[it].x);
    lBt[0][bC + 1][bR[it]] = f2bf(rb[it].y);
    lBt[0][bC + 2][bR[it]] = f2bf(rb[it].z);
    lBt[0][bC + 3][bR[it]] = f2bf(rb[it].w);
  }
  __syncthreads();

  int p = 0;
  for (int k0 = 0; k0 < K; k0 += 32) {
    const bool more = (k0 + 32) < K;
    // issue next chunk's global loads early (overlap with WMMA work)
    if (more) {
      const int kn = k0 + 32;
#pragma unroll
      for (int it = 0; it < 4; ++it)
        ra[it] = *(const float4*)&Ab[(long)(m0 + aR[it]) * K + (kn + aC)];
#pragma unroll
      for (int it = 0; it < 2; ++it)
        rb[it] = *(const float4*)&Bb[(long)(kn + bR[it]) * Nc + (n0 + bC)];
    }

    // Fragments per documented 16-bit layouts:
    // A 16x32: lanes0-15 M=lane; VGPR v<4 -> K=2v(+8*half), v>=4 -> K=2v+8(+8*half)
    // B 32x16: lanes0-15 N=lane; VGPR v<4 -> K=2v(+8*half), v>=4 -> K=2v+8(+8*half)
    FragBF af[2], bf2[2];
#pragma unroll
    for (int t = 0; t < 2; ++t) {
      const int mrow = wm + t * 16 + l16;
      const int ncol = wn + t * 16 + l16;
#pragma unroll
      for (int v = 0; v < 8; ++v) {
        const int ka = (v < 4 ? 2 * v : 2 * v + 8) + 8 * half;
        af[t].u[v]  = *(const unsigned*)&lA[p][mrow][ka];
        const int kb = (v < 4 ? 2 * v : 2 * v + 8) + 8 * half;
        bf2[t].u[v] = *(const unsigned*)&lBt[p][ncol][kb];
      }
    }
#pragma unroll
    for (int i = 0; i < 2; ++i)
#pragma unroll
      for (int j = 0; j < 2; ++j)
        acc[i][j] = __builtin_amdgcn_wmma_f32_16x16x32_bf16(
            false, af[i].v, false, bf2[j].v, (short)0, acc[i][j], false, false);

    if (more) {
      const int q = p ^ 1;
#pragma unroll
      for (int it = 0; it < 4; ++it) {
        uint2 u; u.x = pack_bf2(ra[it].x, ra[it].y); u.y = pack_bf2(ra[it].z, ra[it].w);
        *(uint2*)&lA[q][aR[it]][aC] = u;
      }
#pragma unroll
      for (int it = 0; it < 2; ++it) {
        lBt[q][bC + 0][bR[it]] = f2bf(rb[it].x);
        lBt[q][bC + 1][bR[it]] = f2bf(rb[it].y);
        lBt[q][bC + 2][bR[it]] = f2bf(rb[it].z);
        lBt[q][bC + 3][bR[it]] = f2bf(rb[it].w);
      }
      __syncthreads();   // single barrier per K-chunk (frag loads of p already done)
      p = q;
    }
  }

  // Epilogue: C/D layout: lanes0-15 col=lane, VGPR r -> row = 8*half + r
#pragma unroll
  for (int i = 0; i < 2; ++i)
#pragma unroll
    for (int j = 0; j < 2; ++j)
#pragma unroll
      for (int r = 0; r < 8; ++r) {
        const int row = m0 + wm + i * 16 + half * 8 + r;
        const int col = n0 + wn + j * 16 + l16;
        float v = acc[i][j][r];
        if (mode != MODE_PLAIN) v += bias[col];
        if (mode == MODE_SIGMOID)      v = 1.0f / (1.0f + __expf(-v));
        else if (mode == MODE_RELU)    v = fmaxf(v, 0.0f);
        Cb[(long)row * Nc + col] = v;
      }
}

// ---------------------------------------------------------------------------
// Adjacency normalization: d = rsqrt(max(rowsum_with_unit_diag, 1))
__global__ void __launch_bounds__(256)
rowsum_rsqrt(const float* __restrict__ adj, float* __restrict__ d, int n) {
  __shared__ float sh[256];
  const long row = blockIdx.x;            // b*n + i
  const int  i   = (int)(row % n);
  const float* a = adj + row * (long)n;
  float s = 0.f;
  for (int j = threadIdx.x; j < n; j += 256) s += (j == i) ? 1.0f : a[j];
  const int t = threadIdx.x;
  sh[t] = s; __syncthreads();
  for (int o = 128; o > 0; o >>= 1) { if (t < o) sh[t] += sh[t + o]; __syncthreads(); }
  if (t == 0) d[row] = rsqrtf(fmaxf(sh[0], 1.0f));
}

__global__ void __launch_bounds__(256)
norm_scale(const float* __restrict__ adj, const float* __restrict__ d,
           float* __restrict__ An, int n, long total) {
  const long idx = (long)blockIdx.x * 256 + threadIdx.x;
  if (idx >= total) return;
  const long bn = idx / n; const int j = (int)(idx % n);
  const long b  = bn / n;  const int i = (int)(bn % n);
  const float a = (i == j) ? 1.0f : adj[idx];
  An[idx] = d[b * n + i] * a * d[b * n + j];
}

// ---------------------------------------------------------------------------
// Monte-Carlo gate: V = mean(100 hash-uniforms), gate = sigmoid((V+p-1)/0.1),
// adj_out = adj_in * gate; per-block partials of 0.5*(p-0.5)^2 (deterministic).
__global__ void __launch_bounds__(256)
gate_kernel(const float* __restrict__ p, const float* __restrict__ adj_in,
            float* __restrict__ adj_out, float* __restrict__ part,
            long total, unsigned seed) {
  __shared__ float sh[256];
  const long idx = (long)blockIdx.x * 256 + threadIdx.x;
  float di = 0.f;
  if (idx < total) {
    const float pv = p[idx];
    unsigned st = (unsigned)idx * 2654435761u + seed;
    float sum = 0.f;
#pragma unroll 4
    for (int s = 0; s < 100; ++s) {        // PCG-RXS-M-XS style
      st = st * 747796405u + 2891336453u;
      unsigned w = ((st >> ((st >> 28) + 4u)) ^ st) * 277803737u;
      w = (w >> 22) ^ w;
      sum += (float)(w >> 8) * (1.0f / 16777216.0f);
    }
    const float V = sum * 0.01f;
    const float g = 1.0f / (1.0f + __expf(-(V + pv - 1.0f) * 10.0f));
    adj_out[idx] = adj_in[idx] * g;
    const float e = pv - 0.5f;
    di = 0.5f * e * e;
  }
  const int t = threadIdx.x;
  sh[t] = di; __syncthreads();
  for (int o = 128; o > 0; o >>= 1) { if (t < o) sh[t] += sh[t + o]; __syncthreads(); }
  if (t == 0) part[blockIdx.x] = sh[0];
}

__global__ void __launch_bounds__(256)
reduce_add(const float* __restrict__ part, int cnt, float* __restrict__ info) {
  __shared__ float sh[256];
  float s = 0.f;
  for (int i = threadIdx.x; i < cnt; i += 256) s += part[i];
  const int t = threadIdx.x;
  sh[t] = s; __syncthreads();
  for (int o = 128; o > 0; o >>= 1) { if (t < o) sh[t] += sh[t + o]; __syncthreads(); }
  if (t == 0) info[0] += sh[0];
}

__global__ void init_info(float* info) { info[0] = 0.f; }

// ---------------------------------------------------------------------------
// BatchNorm over (batch, channel) per node i; optional trailing ReLU.
__global__ void __launch_bounds__(256)
bn_kernel(const float* __restrict__ hin, float* __restrict__ hout,
          const float* __restrict__ g, const float* __restrict__ be,
          int n, int C, int postRelu) {
  __shared__ float sh[256], sh2[256];
  const int i = blockIdx.x;
  const int cnt = 32 * C;
  float s = 0.f, s2 = 0.f;
  for (int t = threadIdx.x; t < cnt; t += 256) {
    const int b = t / C, c = t - b * C;
    const float v = hin[((long)b * n + i) * C + c];
    s += v; s2 += v * v;
  }
  const int t = threadIdx.x;
  sh[t] = s; sh2[t] = s2; __syncthreads();
  for (int o = 128; o > 0; o >>= 1) {
    if (t < o) { sh[t] += sh[t + o]; sh2[t] += sh2[t + o]; }
    __syncthreads();
  }
  const float m    = sh[0] / cnt;
  const float var  = sh2[0] / cnt - m * m;
  const float scl  = g[i] * rsqrtf(var + 1e-5f);
  const float shft = be[i] - m * scl;
  for (int tt = threadIdx.x; tt < cnt; tt += 256) {
    const int b = tt / C, c = tt - b * C;
    const long off = ((long)b * n + i) * C + c;
    float v = hin[off] * scl + shft;
    if (postRelu) v = fmaxf(v, 0.f);
    hout[off] = v;
  }
}

// ---------------------------------------------------------------------------
// Top-K pooling pieces
__global__ void __launch_bounds__(256)
vec_norm(const float* __restrict__ w, int len, float* __restrict__ out) {
  __shared__ float sh[256];
  float s = 0.f;
  for (int i = threadIdx.x; i < len; i += 256) { const float v = w[i]; s += v * v; }
  const int t = threadIdx.x;
  sh[t] = s; __syncthreads();
  for (int o = 128; o > 0; o >>= 1) { if (t < o) sh[t] += sh[t + o]; __syncthreads(); }
  if (t == 0) out[0] = sqrtf(sh[0]);
}

__global__ void __launch_bounds__(128)
score_kernel(const float* __restrict__ h, const float* __restrict__ w,
             const float* __restrict__ nrm, int C, float* __restrict__ score) {
  __shared__ float sh[128];
  const long row = blockIdx.x;
  const float* hp = h + row * (long)C;
  float s = 0.f;
  for (int c = threadIdx.x; c < C; c += 128) s += hp[c] * w[c];
  const int t = threadIdx.x;
  sh[t] = s; __syncthreads();
  for (int o = 64; o > 0; o >>= 1) { if (t < o) sh[t] += sh[t + o]; __syncthreads(); }
  if (t == 0) score[row] = tanhf(sh[0] / nrm[0]);
}

// One block per batch; block size == n (power of two). Stable (min-index) ties,
// descending order — matches lax.top_k.
__global__ void topk_select(const float* __restrict__ score, int n, int k,
                            int* __restrict__ idx, float* __restrict__ val) {
  __shared__ float sv[512];
  __shared__ float rv[512];
  __shared__ int   ri[512];
  __shared__ int   win;
  const int b = blockIdx.x, t = threadIdx.x;
  sv[t] = score[(long)b * n + t];
  __syncthreads();
  for (int it = 0; it < k; ++it) {
    rv[t] = sv[t]; ri[t] = t; __syncthreads();
    for (int o = n >> 1; o > 0; o >>= 1) {
      if (t < o) {
        if (rv[t + o] > rv[t] || (rv[t + o] == rv[t] && ri[t + o] < ri[t])) {
          rv[t] = rv[t + o]; ri[t] = ri[t + o];
        }
      }
      __syncthreads();
    }
    if (t == 0) { idx[(long)b * k + it] = ri[0]; val[(long)b * k + it] = rv[0]; win = ri[0]; }
    __syncthreads();
    if (t == win) sv[t] = -3.0e38f;
    __syncthreads();
  }
}

__global__ void __launch_bounds__(256)
gather_pool(const float* __restrict__ h, const int* __restrict__ idx,
            const float* __restrict__ val, float* __restrict__ out,
            int n, int k, int C, long total) {
  const long id = (long)blockIdx.x * 256 + threadIdx.x;
  if (id >= total) return;
  const int  c  = (int)(id % C);
  const long bj = id / C;
  const int  j  = (int)(bj % k);
  const long b  = bj / k;
  const int src = idx[b * k + j];
  out[id] = h[((long)b * n + src) * C + c] * val[b * k + j];
}

// adj_p[b,j1,j2] = ((adj[i1,i2]+adj[i2,i1]) != 0) + (i1==i2)
__global__ void __launch_bounds__(256)
pool_adj(const float* __restrict__ adj, const int* __restrict__ idx,
         float* __restrict__ out, int n, int k, long total) {
  const long id = (long)blockIdx.x * 256 + threadIdx.x;
  if (id >= total) return;
  const int  j2 = (int)(id % k);
  const long bj = id / k;
  const int  j1 = (int)(bj % k);
  const long b  = bj / k;
  const int i1 = idx[b * k + j1], i2 = idx[b * k + j2];
  const float s = adj[((long)b * n + i1) * n + i2] + adj[((long)b * n + i2) * n + i1];
  out[id] = (s != 0.f ? 1.f : 0.f) + (i1 == i2 ? 1.f : 0.f);
}

// ---------------------------------------------------------------------------
// Final tiny linears
__global__ void __launch_bounds__(128)
lin1_kernel(const float* __restrict__ h2, const float* __restrict__ Wl1,
            const float* __restrict__ bl1, float* __restrict__ s1) {
  __shared__ float sh[128];
  const long row = blockIdx.x;                 // b*32 + j
  const float* hp = h2 + row * 128;
  float s = hp[threadIdx.x] * Wl1[threadIdx.x];
  const int t = threadIdx.x;
  sh[t] = s; __syncthreads();
  for (int o = 64; o > 0; o >>= 1) { if (t < o) sh[t] += sh[t + o]; __syncthreads(); }
  if (t == 0) s1[row] = fmaxf(sh[0] + bl1[0], 0.f);
}

__global__ void lin2_kernel(const float* __restrict__ s1, const float* __restrict__ Wl2,
                            const float* __restrict__ bl2, const float* __restrict__ info,
                            float* __restrict__ out) {
  const int t = threadIdx.x;                    // 256 = 32 batches * 8 outputs
  const int b = t >> 3, o = t & 7;
  float s = bl2[o];
  for (int j = 0; j < 32; ++j) s += s1[b * 32 + j] * Wl2[j * 8 + o];
  out[t] = fmaxf(s, 0.f);
  if (t == 0) out[256] = info[0];
}

// ---------------------------------------------------------------------------
extern "C" void kernel_launch(void* const* d_in, const int* in_sizes, int n_in,
                              void* d_out, int out_size, void* d_ws, size_t ws_size,
                              hipStream_t stream) {
  (void)in_sizes; (void)n_in; (void)out_size; (void)ws_size;
  const float* x    = (const float*)d_in[0];
  const float* adj  = (const float*)d_in[1];
  const float* Waa0 = (const float*)d_in[2];
  const float* baa0 = (const float*)d_in[3];
  const float* Waa1 = (const float*)d_in[4];
  const float* baa1 = (const float*)d_in[5];
  const float* W0a  = (const float*)d_in[6];
  const float* b0a  = (const float*)d_in[7];
  const float* g0a  = (const float*)d_in[8];
  const float* be0a = (const float*)d_in[9];
  const float* W0b  = (const float*)d_in[10];
  const float* b0b  = (const float*)d_in[11];
  const float* g0b  = (const float*)d_in[12];
  const float* be0b = (const float*)d_in[13];
  const float* W1a  = (const float*)d_in[14];
  const float* b1a  = (const float*)d_in[15];
  const float* g1a  = (const float*)d_in[16];
  const float* be1a = (const float*)d_in[17];
  const float* W1b  = (const float*)d_in[18];
  const float* b1b  = (const float*)d_in[19];
  const float* g1b  = (const float*)d_in[20];
  const float* be1b = (const float*)d_in[21];
  const float* wp0  = (const float*)d_in[22];
  const float* wp1  = (const float*)d_in[23];
  const float* Wl1  = (const float*)d_in[24];
  const float* bl1  = (const float*)d_in[25];
  const float* Wl2  = (const float*)d_in[26];
  const float* bl2  = (const float*)d_in[27];
  float* out = (float*)d_out;

  const long S_NN = 32L * 512 * 512;      // 8,388,608
  const long S_PP = 32L * 128 * 128;      //   524,288

  char* wsp = (char*)d_ws; size_t off = 0;
  auto alloc = [&](size_t bytes) -> char* {
    char* p = wsp + off; off = (off + bytes + 255) & ~(size_t)255; return p;
  };
  float* INFO  = (float*)alloc(4);
  float* WNRM  = (float*)alloc(4);
  float* D     = (float*)alloc(32L * 512 * 4);
  float* PART  = (float*)alloc(32768L * 4);
  float* AN    = (float*)alloc(S_NN * 4);            // normalized adjacency
  float* XW    = (float*)alloc(S_NN * 4);            // GEMM operand/result pool
  float* G     = (float*)alloc(S_NN * 4);            // GEMM result pool
  float* ADJ1  = (float*)alloc(S_NN * 4);            // gated adjacency step0
  float* HP    = (float*)alloc(32L * 128 * 512 * 4); // pooled features
  float* ADJP  = (float*)alloc(S_PP * 4);
  float* ADJ2  = (float*)alloc(S_PP * 4);
  float* SCORE = (float*)alloc(32L * 512 * 4);
  int*   IDX   = (int*)  alloc(32L * 128 * 4);
  float* TVAL  = (float*)alloc(32L * 128 * 4);
  float* H2    = (float*)alloc(32L * 32 * 128 * 4);
  float* S1    = (float*)alloc(32L * 32 * 4);

  init_info<<<1, 1, 0, stream>>>(INFO);

  // ===================== step 0 (N=512) =====================
  rowsum_rsqrt<<<32 * 512, 256, 0, stream>>>(adj, D, 512);
  norm_scale<<<(int)(S_NN / 256), 256, 0, stream>>>(adj, D, AN, 512, S_NN);
  // XW0 = x @ Waa0        [16384,512] x [512,512]
  gemm_wmma_bf16<<<dim3(8, 128, 1), 256, 0, stream>>>(x, Waa0, XW, 512, 512, 0, 0, 0, nullptr, MODE_PLAIN);
  // p = sigmoid(AN @ XW0 + baa0)   batched M=512,N=512,K=512
  gemm_wmma_bf16<<<dim3(8, 4, 32), 256, 0, stream>>>(AN, XW, G, 512, 512,
      512L * 512, 512L * 512, 512L * 512, baa0, MODE_SIGMOID);
  gate_kernel<<<32768, 256, 0, stream>>>(G, adj, ADJ1, PART, S_NN, 0x9E3779B9u);
  reduce_add<<<1, 256, 0, stream>>>(PART, 32768, INFO);

  rowsum_rsqrt<<<32 * 512, 256, 0, stream>>>(ADJ1, D, 512);
  norm_scale<<<(int)(S_NN / 256), 256, 0, stream>>>(ADJ1, D, AN, 512, S_NN);
  // T = x @ W0a           [16384,64]
  gemm_wmma_bf16<<<dim3(1, 128, 1), 256, 0, stream>>>(x, W0a, XW, 64, 512, 0, 0, 0, nullptr, MODE_PLAIN);
  // H = relu(AN @ T + b0a)   batched M=512,N=64,K=512
  gemm_wmma_bf16<<<dim3(1, 4, 32), 256, 0, stream>>>(AN, XW, G, 64, 512,
      512L * 512, 512L * 64, 512L * 64, b0a, MODE_RELU);
  bn_kernel<<<512, 256, 0, stream>>>(G, XW, g0a, be0a, 512, 64, 0);
  // U = h1 @ W0b          [16384,512] K=64
  gemm_wmma_bf16<<<dim3(8, 128, 1), 256, 0, stream>>>(XW, W0b, G, 512, 64, 0, 0, 0, nullptr, MODE_PLAIN);
  // H2 = relu(AN @ U + b0b)  batched M=512,N=512,K=512
  gemm_wmma_bf16<<<dim3(8, 4, 32), 256, 0, stream>>>(AN, G, XW, 512, 512,
      512L * 512, 512L * 512, 512L * 512, b0b, MODE_RELU);
  bn_kernel<<<512, 256, 0, stream>>>(XW, G, g0b, be0b, 512, 512, 1); // + outer relu

  vec_norm<<<1, 256, 0, stream>>>(wp0, 512, WNRM);
  score_kernel<<<32 * 512, 128, 0, stream>>>(G, wp0, WNRM, 512, SCORE);
  topk_select<<<32, 512, 0, stream>>>(SCORE, 512, 128, IDX, TVAL);
  gather_pool<<<(int)((32L * 128 * 512) / 256), 256, 0, stream>>>(G, IDX, TVAL, HP, 512, 128, 512, 32L * 128 * 512);
  pool_adj<<<(int)(S_PP / 256), 256, 0, stream>>>(ADJ1, IDX, ADJP, 512, 128, S_PP);

  // ===================== step 1 (N=128) =====================
  rowsum_rsqrt<<<32 * 128, 256, 0, stream>>>(ADJP, D, 128);
  norm_scale<<<(int)(S_PP / 256), 256, 0, stream>>>(ADJP, D, AN, 128, S_PP);
  // XW1 = hp @ Waa1       [4096,128] K=512
  gemm_wmma_bf16<<<dim3(2, 32, 1), 256, 0, stream>>>(HP, Waa1, XW, 128, 512, 0, 0, 0, nullptr, MODE_PLAIN);
  // p1 = sigmoid(AN @ XW1 + baa1)   batched M=128,N=128,K=128
  gemm_wmma_bf16<<<dim3(2, 1, 32), 256, 0, stream>>>(AN, XW, G, 128, 128,
      128L * 128, 128L * 128, 128L * 128, baa1, MODE_SIGMOID);
  gate_kernel<<<(int)(S_PP / 256), 256, 0, stream>>>(G, ADJP, ADJ2, PART, S_PP, 0x85EBCA6Bu);
  reduce_add<<<1, 256, 0, stream>>>(PART, (int)(S_PP / 256), INFO);

  rowsum_rsqrt<<<32 * 128, 256, 0, stream>>>(ADJ2, D, 128);
  norm_scale<<<(int)(S_PP / 256), 256, 0, stream>>>(ADJ2, D, AN, 128, S_PP);
  // T1 = hp @ W1a         [4096,64] K=512
  gemm_wmma_bf16<<<dim3(1, 32, 1), 256, 0, stream>>>(HP, W1a, XW, 64, 512, 0, 0, 0, nullptr, MODE_PLAIN);
  // relu(AN @ T1 + b1a)   batched M=128,N=64,K=128
  gemm_wmma_bf16<<<dim3(1, 1, 32), 256, 0, stream>>>(AN, XW, G, 64, 128,
      128L * 128, 128L * 64, 128L * 64, b1a, MODE_RELU);
  bn_kernel<<<128, 256, 0, stream>>>(G, XW, g1a, be1a, 128, 64, 0);
  // U1 = h1p @ W1b        [4096,128] K=64
  gemm_wmma_bf16<<<dim3(2, 32, 1), 256, 0, stream>>>(XW, W1b, G, 128, 64, 0, 0, 0, nullptr, MODE_PLAIN);
  // relu(AN @ U1 + b1b)   batched M=128,N=128,K=128
  gemm_wmma_bf16<<<dim3(2, 1, 32), 256, 0, stream>>>(AN, G, XW, 128, 128,
      128L * 128, 128L * 128, 128L * 128, b1b, MODE_RELU);
  bn_kernel<<<128, 256, 0, stream>>>(XW, G, g1b, be1b, 128, 128, 1); // + outer relu

  vec_norm<<<1, 256, 0, stream>>>(wp1, 128, WNRM);
  score_kernel<<<32 * 128, 128, 0, stream>>>(G, wp1, WNRM, 128, SCORE);
  topk_select<<<32, 128, 0, stream>>>(SCORE, 128, 32, IDX, TVAL);
  gather_pool<<<(int)((32L * 32 * 128) / 256), 256, 0, stream>>>(G, IDX, TVAL, H2, 128, 32, 128, 32L * 32 * 128);

  lin1_kernel<<<32 * 32, 128, 0, stream>>>(H2, Wl1, bl1, S1);
  lin2_kernel<<<1, 256, 0, stream>>>(S1, Wl2, bl2, INFO, out);
}
